// EISANImodel_78632261255731
// MI455X (gfx1250) — 16-bit and 8-bit matrix operand order repaired and checked by reading, compiled, not run on hardware
//
#include <hip/hip_runtime.h>
#include <stdint.h>

typedef __attribute__((ext_vector_type(8)))  int      v8i;
typedef __attribute__((ext_vector_type(8)))  float    v8f;
typedef __attribute__((ext_vector_type(16))) _Float16 v16h;

#define BATCH       512
#define FEAT        256
#define HIDDEN      4096
#define CLASSES     1000
#define ENC         2048
#define ACAT_STRIDE 12288
#define THRESH_I    4

__global__ __launch_bounds__(256) void encode_kernel(
    const float* __restrict__ x, uint8_t* __restrict__ a0)
{
  const int idx = blockIdx.x * 256 + threadIdx.x;
  if (idx >= BATCH * FEAT) return;
  float v = x[idx];
  v = fminf(fmaxf(v, 0.0f), 1.0f);
  const int level = (int)rintf(v * 255.0f);
  const int gray  = level ^ (level >> 1);
  uint32_t lo = 0u, hi = 0u;
#pragma unroll
  for (int i = 0; i < 4; ++i) lo |= (uint32_t)((gray >> i) & 1) << (8 * i);
#pragma unroll
  for (int i = 0; i < 4; ++i) hi |= (uint32_t)((gray >> (4 + i)) & 1) << (8 * i);
  uint32_t* dst = (uint32_t*)(a0 + (size_t)idx * 8);
  dst[0] = lo;
  dst[1] = hi;
}

__global__ __launch_bounds__(256) void zero_kernel(float* __restrict__ p, int n)
{
  const int i = blockIdx.x * 256 + threadIdx.x;
  if (i < n) p[i] = 0.0f;
}

__global__ __launch_bounds__(256) void layer_kernel(
    const uint8_t* __restrict__ aprev, int aprev_stride, int K,
    const float* __restrict__ W,
    uint8_t* __restrict__ anext, int anext_stride)
{
  __shared__ alignas(32) uint32_t bTile[2][256];

  const int tid  = threadIdx.x;
  const int wave = tid >> 5;
  const int lane = tid & 31;
  const int half = lane >> 4;
  const int nloc = lane & 15;
  const int ntile = blockIdx.x << 4;

  const int c_lane = tid >> 3;
  const int c_j    = tid & 7;
  const int c_n    = c_lane & 15;
  const int c_half = c_lane >> 4;
  const int c_koff = ((c_j >> 2) << 5) + (c_half << 4) + ((c_j & 3) << 2);
  const float* wrow = W + (size_t)(ntile + c_n) * (size_t)K + c_koff;

  v8i acc[4];
#pragma unroll
  for (int b = 0; b < 4; ++b)
#pragma unroll
    for (int j = 0; j < 8; ++j) acc[b][j] = 0;

  const int mbase = wave << 6;

  float4 wv = *(const float4*)(wrow);
  int buf = 0;

  for (int k0 = 0; k0 < K; k0 += 64) {
    const int i0 = (int)wv.x, i1 = (int)wv.y, i2 = (int)wv.z, i3 = (int)wv.w;
    const uint32_t packed =
        (uint32_t)(i0 & 0xFF)        | ((uint32_t)(i1 & 0xFF) << 8) |
        ((uint32_t)(i2 & 0xFF) << 16) | ((uint32_t)(i3 & 0xFF) << 24);

    const int knext = (k0 + 64 < K) ? (k0 + 64) : k0;
    const float4 wnext = *(const float4*)(wrow + knext);
    __builtin_prefetch(wrow + k0 + 256, 0, 3);

    bTile[buf][tid] = packed;
    __syncthreads();

    const v8i bop = *(const v8i*)&bTile[buf][lane << 3];

#pragma unroll
    for (int bt = 0; bt < 4; ++bt) {
      const uint32_t* arow = (const uint32_t*)(aprev +
          (size_t)(mbase + (bt << 4) + nloc) * (size_t)aprev_stride + k0);
      v8i aop;
#pragma unroll
      for (int j = 0; j < 8; ++j) {
        const int dw = ((j >> 2) << 3) + (half << 2) + (j & 3);
        aop[j] = (int)arow[dw];
      }
      acc[bt] = __builtin_amdgcn_wmma_i32_16x16x64_iu8(
           false, aop,  true, bop, acc[bt],
           false,  false);
    }

    wv = wnext;
    buf ^= 1;
  }

#pragma unroll
  for (int bt = 0; bt < 4; ++bt)
#pragma unroll
    for (int r = 0; r < 8; ++r) {
      const int m = mbase + (bt << 4) + (half << 3) + r;
      anext[(size_t)m * (size_t)anext_stride + ntile + nloc] =
          (acc[bt][r] >= THRESH_I) ? (uint8_t)1 : (uint8_t)0;
    }
}

__global__ __launch_bounds__(256) void outgemm_kernel(
    const uint8_t* __restrict__ acat,
    const float* __restrict__ wo,
    float* __restrict__ out)
{
  __shared__ alignas(32) uint32_t bTile[2][256];

  const int tid  = threadIdx.x;
  const int wave = tid >> 5;
  const int lane = tid & 31;
  const int half = lane >> 4;
  const int nloc = lane & 15;
  const int ntile = blockIdx.x << 4;
  const int kbase = blockIdx.y << 10;

  const int c_lane = tid >> 3;
  const int c_j    = tid & 7;
  int c_col = ntile + (c_lane & 15);
  if (c_col > CLASSES - 1) c_col = CLASSES - 1;
  const int c_half = c_lane >> 4;
  const float* wp = wo + (size_t)(kbase + (c_half << 4) + (c_j << 1)) * CLASSES + c_col;

  v8f acc[4];
#pragma unroll
  for (int b = 0; b < 4; ++b)
#pragma unroll
    for (int j = 0; j < 8; ++j) acc[b][j] = 0.0f;

  const int mbase = wave << 6;

  float f0 = wp[0];
  float f1 = wp[CLASSES];
  int buf = 0;

  for (int k0 = 0; k0 < 1024; k0 += 32) {
    union { _Float16 h[2]; uint32_t u; } cv;
    cv.h[0] = (_Float16)f0;
    cv.h[1] = (_Float16)f1;

    const int knext = (k0 + 32 < 1024) ? (k0 + 32) : k0;
    const float f0n = wp[(size_t)knext * CLASSES];
    const float f1n = wp[(size_t)knext * CLASSES + CLASSES];
    __builtin_prefetch(wp + (size_t)(k0 + 64) * CLASSES, 0, 1);

    bTile[buf][tid] = cv.u;
    __syncthreads();

    union { uint32_t u[8]; v16h h; } bv;
#pragma unroll
    for (int j = 0; j < 8; ++j) bv.u[j] = bTile[buf][(lane << 3) + j];

#pragma unroll
    for (int bt = 0; bt < 4; ++bt) {
      const uint16_t* arow = (const uint16_t*)(acat +
          (size_t)(mbase + (bt << 4) + nloc) * ACAT_STRIDE + kbase + k0);
      union { uint32_t u[8]; v16h h; } av;
#pragma unroll
      for (int j = 0; j < 8; ++j) {
        const int i16 = (half << 3) + j;
        const uint32_t v = arow[i16];
        av.u[j] = (v & 1u) * 0x3C00u + (v & 0x100u) * 0x3C0000u;
      }
      acc[bt] = __builtin_amdgcn_wmma_f32_16x16x32_f16(
           false, av.h,  false, bv.h,
           (short)0, acc[bt],  false,  false);
    }

    f0 = f0n;
    f1 = f1n;
    buf ^= 1;
  }

  const int ncol = ntile + nloc;
  if (ncol < CLASSES) {
#pragma unroll
    for (int bt = 0; bt < 4; ++bt)
#pragma unroll
      for (int r = 0; r < 8; ++r) {
        const int m = mbase + (bt << 4) + (half << 3) + r;
        atomicAdd(out + (size_t)m * CLASSES + ncol, acc[bt][r]);
      }
  }
}

extern "C" void kernel_launch(void* const* d_in, const int* in_sizes, int n_in,
                              void* d_out, int out_size, void* d_ws, size_t ws_size,
                              hipStream_t stream) {
  (void)in_sizes; (void)n_in; (void)out_size; (void)ws_size;
  const float* x     = (const float*)d_in[0];
  const float* W0    = (const float*)d_in[1];
  const float* W1    = (const float*)d_in[2];
  const float* W2    = (const float*)d_in[3];
  const float* out_w = (const float*)d_in[4];
  float* out = (float*)d_out;

  uint8_t* A0   = (uint8_t*)d_ws;
  uint8_t* Acat = A0 + (size_t)BATCH * ENC;

  encode_kernel<<<(BATCH * FEAT + 255) / 256, 256, 0, stream>>>(x, A0);
  zero_kernel<<<(BATCH * CLASSES + 255) / 256, 256, 0, stream>>>(out, BATCH * CLASSES);

  layer_kernel<<<HIDDEN / 16, 256, 0, stream>>>(A0, ENC, ENC, W0,
                                                Acat + 0, ACAT_STRIDE);
  layer_kernel<<<HIDDEN / 16, 256, 0, stream>>>(Acat + 0, ACAT_STRIDE, HIDDEN, W1,
                                                Acat + HIDDEN, ACAT_STRIDE);
  layer_kernel<<<HIDDEN / 16, 256, 0, stream>>>(Acat + HIDDEN, ACAT_STRIDE, HIDDEN, W2,
                                                Acat + 2 * HIDDEN, ACAT_STRIDE);

  outgemm_kernel<<<dim3(63, 12), 256, 0, stream>>>(Acat, out_w, out);
}
